// TAttention_20340965114203
// MI455X (gfx1250) — matrix operand order repaired and checked by reading, not compiled, not run on hardware
//
#include <hip/hip_runtime.h>
#include <hip/hip_bf16.h>

typedef __attribute__((ext_vector_type(16))) _Float16 v16h;
typedef __attribute__((ext_vector_type(8)))  float    v8f;

#define B_  16
#define C_  32
#define N_  1024
#define T_  128
#define R_  10

// ---------------------------------------------------------------------------
// Kernel 1: k[b][n][t] = sum_c alpha[c] * x[b][c][n][t]
// Grid: B*N blocks, T threads. Coalesced 512B row reads per c, coalesced write.
// Pure streaming: 256 MB read @ 23.3 TB/s ~= 11 us.
// ---------------------------------------------------------------------------
__global__ void __launch_bounds__(T_)
k_reduce_c(const float* __restrict__ x, const float* __restrict__ alpha,
           float* __restrict__ k) {
    const int bn = blockIdx.x;            // b*N + n
    const int t  = threadIdx.x;           // 0..127
    const int b  = bn >> 10;              // / N_
    const int n  = bn & (N_ - 1);
    const float* xp = x + (((size_t)b * C_) * N_ + n) * T_ + t;
    float acc = 0.f;
#pragma unroll
    for (int c = 0; c < C_; ++c)
        acc += alpha[c] * xp[(size_t)c * N_ * T_];
    k[(size_t)bn * T_ + t] = acc;
}

// ---------------------------------------------------------------------------
// Kernel 2: kw1[b][t][r] = sum_n k[b][n][t]*W1[r][n]  (and same for W2->qw2)
// Grid: B*T blocks, 1 wave; lanes 0..9 -> W1, lanes 10..19 -> W2.
// k is 8 MB -> L2-resident; lane loads of k broadcast (same address).
// ---------------------------------------------------------------------------
__global__ void __launch_bounds__(32)
kw_gemv(const float* __restrict__ k, const float* __restrict__ W1,
        const float* __restrict__ W2, float* __restrict__ kw1,
        float* __restrict__ qw2) {
    const int bt  = blockIdx.x;           // b*T + t
    const int b   = bt >> 7;
    const int t   = bt & (T_ - 1);
    const int tid = threadIdx.x;
    if (tid >= 2 * R_) return;
    const bool  first = (tid < R_);
    const int   r     = first ? tid : tid - R_;
    const float* wp = (first ? W1 : W2) + (size_t)r * N_;
    const float* kp = k + ((size_t)b * N_) * T_ + t;      // stride T_ over n
    float a0 = 0.f, a1 = 0.f, a2 = 0.f, a3 = 0.f;
    for (int n = 0; n < N_; n += 4) {
        a0 += kp[(size_t)(n + 0) * T_] * wp[n + 0];
        a1 += kp[(size_t)(n + 1) * T_] * wp[n + 1];
        a2 += kp[(size_t)(n + 2) * T_] * wp[n + 2];
        a3 += kp[(size_t)(n + 3) * T_] * wp[n + 3];
    }
    float acc = (a0 + a1) + (a2 + a3);
    (first ? kw1 : qw2)[(size_t)bt * R_ + r] = acc;
}

// ---------------------------------------------------------------------------
// Kernel 3: scores[b][t][s] = sum_r kw1[b][t][r]*qw2[b][s][r]; softmax over s;
// store att in f16 (row-major [b][t][s]) for the WMMA B-fragment loads.
// Grid: B*T blocks, T threads (one score per thread), LDS tree reductions.
// ---------------------------------------------------------------------------
__global__ void __launch_bounds__(T_)
scores_softmax(const float* __restrict__ kw1, const float* __restrict__ qw2,
               _Float16* __restrict__ att) {
    const int bt = blockIdx.x;            // b*T + t
    const int b  = bt >> 7;
    const int s  = threadIdx.x;           // 0..127
    __shared__ float red[T_];

    const float* krow = kw1 + (size_t)bt * R_;
    const float* qrow = qw2 + ((size_t)b * T_ + s) * R_;
    float sc = 0.f;
#pragma unroll
    for (int r = 0; r < R_; ++r) sc += krow[r] * qrow[r];

    // row max
    red[s] = sc;
    __syncthreads();
    for (int off = T_ / 2; off > 0; off >>= 1) {
        if (s < off) red[s] = fmaxf(red[s], red[s + off]);
        __syncthreads();
    }
    const float mx = red[0];
    __syncthreads();

    // exp + sum
    const float e = expf(sc - mx);
    red[s] = e;
    __syncthreads();
    for (int off = T_ / 2; off > 0; off >>= 1) {
        if (s < off) red[s] += red[s + off];
        __syncthreads();
    }
    att[(size_t)bt * T_ + s] = (_Float16)(e / red[0]);
}

// ---------------------------------------------------------------------------
// Kernel 4: out[b][c][n][t] = sum_s att[b][t][s] * x[b][c][n][s]
// Per (b,c): Out[1024x128] = X[1024x128] * Att^T[128x128] via
// V_WMMA_F32_16X16X32_F16. One wave = one 16-row n-tile x all 8 t-tiles,
// K (=s) walked in 4 chunks of 32. Fragment layouts used here (contiguous k, not ISA table order):
//   A 16x32 f16 : lane L -> row M=L&15, halves = K k0..k0+15,
//                 k0 = (L<16 ? 0 : 16)
//   B 32x16 f16 : lane L -> col N=L&15, halves = K k0..k0+15, k0=(L<16?0:16)
//                 == 16 contiguous halves of row-major att[b][t0+N][s]
//   C/D 16x16 f32: vgpr r -> row r+(L>=16?8:0), col L&15
// ---------------------------------------------------------------------------
__global__ void __launch_bounds__(128)
att_apply_wmma(const float* __restrict__ x, const _Float16* __restrict__ att,
               float* __restrict__ out) {
    const int wave  = blockIdx.x * (blockDim.x >> 5) + (threadIdx.x >> 5);
    const int lane  = threadIdx.x & 31;
    const int ntile = wave & 63;                 // N_/16 = 64
    const int c     = (wave >> 6) & 31;
    const int b     = wave >> 11;
    const int m     = lane & 15;
    const bool hi   = lane >= 16;
    const int n0    = ntile << 4;
    const int kb    = hi ? 8 : 0;
    const int k0    = hi ? 16 : 0;

    const float*    xrow  = x   + (((size_t)b * C_ + c) * N_ + n0 + m) * T_;
    const _Float16* abase = att + ((size_t)b * T_) * T_;   // [t][s] row-major

    v8f acc[8] = {};

#pragma unroll
    for (int ks = 0; ks < 4; ++ks) {
        const int s0 = ks * 32;
        // A fragment: 16 f32 -> f16 from the contiguous x row (all 16B aligned)
        const float4 f0 = *(const float4*)(xrow + s0 + k0);
        const float4 f1 = *(const float4*)(xrow + s0 + k0 + 4);
        const float4 f2 = *(const float4*)(xrow + s0 + k0 + 8);
        const float4 f3 = *(const float4*)(xrow + s0 + k0 + 12);
        v16h a;
        a[0]  = (_Float16)f0.x; a[1]  = (_Float16)f0.y;
        a[2]  = (_Float16)f0.z; a[3]  = (_Float16)f0.w;
        a[4]  = (_Float16)f1.x; a[5]  = (_Float16)f1.y;
        a[6]  = (_Float16)f1.z; a[7]  = (_Float16)f1.w;
        a[8]  = (_Float16)f2.x; a[9]  = (_Float16)f2.y;
        a[10] = (_Float16)f2.z; a[11] = (_Float16)f2.w;
        a[12] = (_Float16)f3.x; a[13] = (_Float16)f3.y;
        a[14] = (_Float16)f3.z; a[15] = (_Float16)f3.w;

#pragma unroll
        for (int tt = 0; tt < 8; ++tt) {
            // B fragment: 16 contiguous halves, 32B aligned (att is L2-resident)
            const v16h bf =
                *(const v16h*)(abase + ((size_t)(tt * 16 + m)) * T_ + s0 + k0);
            acc[tt] = __builtin_amdgcn_wmma_f32_16x16x32_f16(
                /*neg_a=*/false, a, /*neg_b=*/false, bf,
                /*c_mod=*/(short)0, acc[tt],
                /*reuse_a=*/false, /*reuse_b=*/false);
        }
    }

    // Store: vgpr r -> n-row n0 + r + (hi?8:0), lane -> t-col t0 + m
    float* obase = out + (((size_t)b * C_ + c) * N_ + n0 + (hi ? 8 : 0)) * T_ + m;
#pragma unroll
    for (int tt = 0; tt < 8; ++tt) {
#pragma unroll
        for (int r = 0; r < 8; ++r) {
            obase[(size_t)r * T_ + tt * 16] = acc[tt][r];
        }
    }
}

// ---------------------------------------------------------------------------
// Host launcher. Workspace layout (all offsets 256B aligned):
//   k    f32 [B][N][T]   @ 0         (8,388,608 B)
//   kw1  f32 [B][T][R]   @ 8388608   (   81,920 B)
//   qw2  f32 [B][T][R]   @ 8470528   (   81,920 B)
//   att  f16 [B][T][T]   @ 8552448   (  524,288 B)   total ~9.1 MB
// ---------------------------------------------------------------------------
extern "C" void kernel_launch(void* const* d_in, const int* in_sizes, int n_in,
                              void* d_out, int out_size, void* d_ws, size_t ws_size,
                              hipStream_t stream) {
    const float* x     = (const float*)d_in[0];
    const float* W1    = (const float*)d_in[1];
    const float* W2    = (const float*)d_in[2];
    const float* alpha = (const float*)d_in[3];
    float* out = (float*)d_out;

    char* ws = (char*)d_ws;
    float*    k   = (float*)(ws);
    float*    kw1 = (float*)(ws + 8388608);
    float*    qw2 = (float*)(ws + 8470528);
    _Float16* att = (_Float16*)(ws + 8552448);

    k_reduce_c    <<<B_ * N_, T_, 0, stream>>>(x, alpha, k);
    kw_gemv       <<<B_ * T_, 32, 0, stream>>>(k, W1, W2, kw1, qw2);
    scores_softmax<<<B_ * T_, T_, 0, stream>>>(kw1, qw2, att);
    // B*C*(N/16) = 32768 waves, 4 waves (128 threads) per block
    att_apply_wmma<<<(B_ * C_ * (N_ / 16)) / 4, 128, 0, stream>>>(x, att, out);
}
